// Head_18038862643604
// MI455X (gfx1250) — matrix operand order repaired and checked by reading, not compiled, not run on hardware
//
#include <hip/hip_runtime.h>
#include <hip/hip_bf16.h>

typedef __attribute__((ext_vector_type(16))) _Float16 v16h;
typedef __attribute__((ext_vector_type(8)))  _Float16 v8h;
typedef __attribute__((ext_vector_type(4)))  _Float16 v4h;
typedef __attribute__((ext_vector_type(8)))  float    v8f;

#define CTX   32
#define NDIM  64
#define HSZ   64
#define WAVES_PER_BLOCK 8
#define BLOCK (WAVES_PER_BLOCK * 32)
#define WAVE_SCRATCH 16384                  // bytes of private LDS per wave
#define SMEM_BYTES (3*64*64*2 + WAVES_PER_BLOCK*WAVE_SCRATCH)  // 24KB weights + 128KB

static __device__ __forceinline__ v8f wmma_f32_f16(v16h a, v16h b, v8f c) {
  // (neg_a, A, neg_b, B, c_mod, C, reuse_a, reuse_b)
  return __builtin_amdgcn_wmma_f32_16x16x32_f16(false, a, false, b, (short)0, c,
                                                false, false);
}

// A-matrix fragment (16-bit, 16x32): lane holds row m0+(lane&15);
// halves 0..7 = K at k0+(lane>>4)*8, halves 8..15 at +16. Two ds_load_b128.
static __device__ __forceinline__ v16h fragA(const _Float16* m, int ld, int m0,
                                             int k0, int lane) {
  int r  = m0 + (lane & 15);
  int kb = k0 + ((lane >> 4) << 3);
  const _Float16* p = m + r * ld + kb;
  v8h lo = *(const v8h*)p;
  v8h hi = *(const v8h*)(p + 16);
  return __builtin_shufflevector(lo, hi, 0,1,2,3,4,5,6,7,8,9,10,11,12,13,14,15);
}

// B-matrix fragment (16-bit, 32x16): lane holds column n0+(lane&15);
// halves 0..7 = K at k0+(lane>>4)*8, halves 8..15 at +16. mT stored [n][k].
static __device__ __forceinline__ v16h fragB(const _Float16* mT, int ld, int n0,
                                             int k0, int lane) {
  int n  = n0 + (lane & 15);
  int kb = k0 + ((lane >> 4) << 3);
  const _Float16* p = mT + n * ld + kb;
  v8h lo = *(const v8h*)p;
  v8h hi = *(const v8h*)(p + 16);
  return __builtin_shufflevector(lo, hi, 0,1,2,3,4,5,6,7,8,9,10,11,12,13,14,15);
}

__global__ void __launch_bounds__(BLOCK)
head_attn_kernel(const float* __restrict__ x,  const float* __restrict__ Wk,
                 const float* __restrict__ Wq, const float* __restrict__ Wv,
                 float* __restrict__ out) {
  extern __shared__ char smem[];
  _Float16* wqT = (_Float16*)smem;            // [h][d] f16 (transposed -> K-contig)
  _Float16* wkT = wqT + 64 * 64;
  _Float16* wvT = wkT + 64 * 64;
  char* wbase = smem + 3 * 64 * 64 * 2;

  const int lane = threadIdx.x & 31;
  const int wave = threadIdx.x >> 5;
  char* my = wbase + wave * WAVE_SCRATCH;     // per-wave private scratch
  _Float16* xh = (_Float16*)my;               // [32][64] f16 x tile
  _Float16* qh = (_Float16*)(my + 4096);      // [32][64] f16 q
  _Float16* kh = (_Float16*)(my + 8192);      // [32][64] f16 k (rows K-contig for B)
  _Float16* vT = (_Float16*)(my + 12288);     // [64][32] f16 v transposed
  float*    sS = (float*)my;                  // [32][32] f32 logits (overlays xh)
  _Float16* ph = (_Float16*)(my + 4096);      // [32][32] f16 probs  (overlays qh)

  // ---- cooperative weight load: fp32 global -> f16 transposed LDS (once) ----
  for (int i = threadIdx.x; i < 64 * 64; i += BLOCK) {
    int d = i >> 6, h = i & 63;               // W is [d_in][h_out] row-major
    wqT[h * 64 + d] = (_Float16)Wq[i];
    wkT[h * 64 + d] = (_Float16)Wk[i];
    wvT[h * 64 + d] = (_Float16)Wv[i];
  }
  __syncthreads();

  const int b = blockIdx.x * WAVES_PER_BLOCK + wave;
  const float* xb = x + (size_t)b * (CTX * NDIM);

  // ---- load x tile (8KB fp32, coalesced b128) -> f16 LDS ----
  const float4* xb4 = (const float4*)xb;
#pragma unroll
  for (int i = 0; i < (CTX * NDIM) / (4 * 32); ++i) {
    int idx = lane + i * 32;
    float4 f = xb4[idx];
    v4h hv = {(_Float16)f.x, (_Float16)f.y, (_Float16)f.z, (_Float16)f.w};
    *(v4h*)(xh + 4 * idx) = hv;
  }

  // ---- projections: q = x@Wq, k = x@Wk, v = x@Wv (A frags shared) ----
#pragma unroll
  for (int mt = 0; mt < 2; ++mt) {
    v16h a0 = fragA(xh, 64, mt * 16, 0, lane);
    v16h a1 = fragA(xh, 64, mt * 16, 32, lane);
#pragma unroll
    for (int nt = 0; nt < 4; ++nt) {
      v8f cq = {}, ck = {}, cv = {};
      cq = wmma_f32_f16(a0, fragB(wqT, 64, nt * 16, 0, lane), cq);
      cq = wmma_f32_f16(a1, fragB(wqT, 64, nt * 16, 32, lane), cq);
      ck = wmma_f32_f16(a0, fragB(wkT, 64, nt * 16, 0, lane), ck);
      ck = wmma_f32_f16(a1, fragB(wkT, 64, nt * 16, 32, lane), ck);
      cv = wmma_f32_f16(a0, fragB(wvT, 64, nt * 16, 0, lane), cv);
      cv = wmma_f32_f16(a1, fragB(wvT, 64, nt * 16, 32, lane), cv);
      int n  = nt * 16 + (lane & 15);
      int mb = mt * 16 + ((lane >> 4) << 3);
#pragma unroll
      for (int e = 0; e < 8; ++e) {
        qh[(mb + e) * 64 + n] = (_Float16)cq[e];
        kh[(mb + e) * 64 + n] = (_Float16)ck[e];
        vT[n * 32 + (mb + e)] = (_Float16)cv[e];   // transpose v for PV B-frags
      }
    }
  }

  // ---- S = q @ k^T  (A = q row-major, B = k rows are K(h)-contiguous) ----
#pragma unroll
  for (int mt = 0; mt < 2; ++mt) {
    v16h a0 = fragA(qh, 64, mt * 16, 0, lane);
    v16h a1 = fragA(qh, 64, mt * 16, 32, lane);
#pragma unroll
    for (int nt = 0; nt < 2; ++nt) {
      v8f c = {};
      c = wmma_f32_f16(a0, fragB(kh, 64, nt * 16, 0, lane), c);
      c = wmma_f32_f16(a1, fragB(kh, 64, nt * 16, 32, lane), c);
      int n  = nt * 16 + (lane & 15);
      int mb = mt * 16 + ((lane >> 4) << 3);
#pragma unroll
      for (int e = 0; e < 8; ++e) sS[(mb + e) * 32 + n] = c[e];  // fp32 logits
    }
  }

  // ---- causal softmax in fp32: lane r owns row r (T == wave32) ----
  {
    const int row = lane;
    float vv[32];
    float mx = -1e30f;
#pragma unroll
    for (int s = 0; s < 32; ++s) {
      float t = sS[row * 32 + s] * 8.0f;      // * sqrt(head_size), per reference
      t = (s <= row) ? t : -1e30f;
      vv[s] = t;
      mx = fmaxf(mx, t);
    }
    float sum = 0.f;
#pragma unroll
    for (int s = 0; s < 32; ++s) {
      float e = __expf(vv[s] - mx);
      e = (s <= row) ? e : 0.f;
      vv[s] = e;
      sum += e;
    }
    float inv = 1.0f / sum;
#pragma unroll
    for (int s = 0; s < 32; ++s)
      ph[row * 32 + s] = (_Float16)(vv[s] * inv);
  }

  // ---- O = P @ V  (K=32 in one WMMA step); store fp32 to global ----
  float* ob = out + (size_t)b * (CTX * HSZ);
#pragma unroll
  for (int mt = 0; mt < 2; ++mt) {
    v16h a = fragA(ph, 32, mt * 16, 0, lane);
#pragma unroll
    for (int nt = 0; nt < 4; ++nt) {
      v8f c = {};
      c = wmma_f32_f16(a, fragB(vT, 32, nt * 16, 0, lane), c);
      int n  = nt * 16 + (lane & 15);
      int mb = mt * 16 + ((lane >> 4) << 3);
#pragma unroll
      for (int e = 0; e < 8; ++e)
        ob[(size_t)(mb + e) * 64 + n] = c[e];
    }
  }
}

extern "C" void kernel_launch(void* const* d_in, const int* in_sizes, int n_in,
                              void* d_out, int out_size, void* d_ws, size_t ws_size,
                              hipStream_t stream) {
  const float* x  = (const float*)d_in[0];
  const float* Wk = (const float*)d_in[1];
  const float* Wq = (const float*)d_in[2];
  const float* Wv = (const float*)d_in[3];
  float* out = (float*)d_out;

  int batch = in_sizes[0] / (CTX * NDIM);          // 16384
  int grid  = batch / WAVES_PER_BLOCK;             // 2048 (exact)
  hipLaunchKernelGGL(head_attn_kernel, dim3(grid), dim3(BLOCK), SMEM_BYTES,
                     stream, x, Wk, Wq, Wv, out);
}
